// SimpleVQ_87385404604677
// MI455X (gfx1250) — compile-verified, matrix operand order checked by reading, not run on hardware
//
#include <hip/hip_runtime.h>
#include <hip/hip_bf16.h>
#include <math.h>

// ---------------------------------------------------------------------------
// SimpleVQ for MI455X (gfx1250, wave32, WMMA + async LDS copies).
//   d[m,n] = |v_m|^2 - 2 * <v_m, c_n> + |c_n|^2
//   cross term: v_wmma_f32_16x16x32_f16 ; codebook staged to LDS with
//   global_load_async_to_lds_b128 (ASYNCcnt), double-buffered.
// ---------------------------------------------------------------------------

typedef __attribute__((ext_vector_type(16))) _Float16 v16h;
typedef __attribute__((ext_vector_type(8)))  _Float16 v8h;
typedef __attribute__((ext_vector_type(4)))  _Float16 v4h;
typedef __attribute__((ext_vector_type(8)))  float    v8f;

#define S_CODES 1024
#define K_DIM   128
#define N_ROWS  65536          // B*H*L = 8*1*8192
#define ROWS_PER_BLOCK 64
#define AH_PITCH 136           // 128 + 8 halfs pad -> 272B row stride (bank skew)
#define CHUNK_CODES 64         // codes staged per async chunk
#define N_CHUNKS (S_CODES / CHUNK_CODES)   // 16
#define SEGS_PER_THREAD 8      // 64 rows * 16 segs / 128 threads

// ---------------------------------------------------------------------------
// Kernel 1: materialize codebook (sinusoid PE -> rmsnorm -> *128^-1/4):
// f32 copy (exact gather), f16 copy (WMMA B), per-code squared norms.
// ---------------------------------------------------------------------------
__global__ __launch_bounds__(128) void vq_codebook_kernel(
    float* __restrict__ cb32, _Float16* __restrict__ cb16,
    float* __restrict__ cnorm)
{
    __shared__ float red[128];
    const int s = blockIdx.x;
    const int d = threadIdx.x;

    const float jj  = (float)(d & 63);
    const float inv = expf(-(2.0f * jj / 128.0f) * logf(100000.0f));
    const float pre = (float)s * inv;
    const float e   = (d < 64) ? sinf(pre) : cosf(pre);

    red[d] = e * e;
    __syncthreads();
    for (int off = 64; off >= 1; off >>= 1) {
        if (d < off) red[d] += red[d + off];
        __syncthreads();
    }
    const float mean = red[0] * (1.0f / 128.0f);
    const float r    = rsqrtf(1e-6f + mean);
    const float y    = 0.29730177875068026f * e * r;   // 128^(-1/4) * rmsnorm

    cb32[s * K_DIM + d] = y;
    cb16[s * K_DIM + d] = (_Float16)y;

    __syncthreads();
    red[d] = y * y;
    __syncthreads();
    for (int off = 64; off >= 1; off >>= 1) {
        if (d < off) red[d] += red[d + off];
        __syncthreads();
    }
    if (d == 0) cnorm[s] = red[0];
}

// ---------------------------------------------------------------------------
// Kernel 2: main VQ. 128 threads = 4 waves, 64 rows per block. Codebook is
// streamed block-wide through double-buffered LDS chunks via async copies;
// each wave owns a 16-row M-subtile and runs 16 WMMAs per chunk.
// ---------------------------------------------------------------------------
__global__ __launch_bounds__(128) void vq_main_kernel(
    const float*     __restrict__ vecs,
    const _Float16*  __restrict__ cb16,
    const float*     __restrict__ cb32,
    const float*     __restrict__ cnorm,
    const float*     __restrict__ mask,
    float*           __restrict__ hat,
    int*             __restrict__ zout,
    float*           __restrict__ eout,
    float*           __restrict__ partials)
{
    __shared__ _Float16 aH[ROWS_PER_BLOCK][AH_PITCH];
    __shared__ _Float16 bS[2][CHUNK_CODES][AH_PITCH];
    __shared__ float vnormS[ROWS_PER_BLOCK];
    __shared__ int   zsh[ROWS_PER_BLOCK];
    __shared__ float errsh[ROWS_PER_BLOCK];
    __shared__ float red[128];

    const int  tid   = threadIdx.x;
    const int  w     = tid >> 5;
    const int  lane  = tid & 31;
    const int  hhalf = lane >> 4;
    const int  l16   = lane & 15;
    const long rowbase = (long)blockIdx.x * ROWS_PER_BLOCK;

    // LDS byte offsets of the two B staging buffers (low 32 bits of the
    // generic pointer are the LDS offset on gfx1250)
    const uint32_t bbase   = (uint32_t)(uintptr_t)&bS[0][0][0];
    const uint32_t bstride = (uint32_t)(CHUNK_CODES * AH_PITCH * sizeof(_Float16));
    const uint64_t cb16b   = (uint64_t)(uintptr_t)cb16;

    // ---- Phase A: stage 64 rows of vecs to LDS as f16; exact f32 row norms
    for (int rr = 0; rr < 16; ++rr) {
        const int lrow = w * 16 + rr;
        const float4 v4 =
            *(const float4*)(vecs + (rowbase + lrow) * K_DIM + lane * 4);
        float ss = v4.x * v4.x + v4.y * v4.y + v4.z * v4.z + v4.w * v4.w;
        #pragma unroll
        for (int m = 16; m >= 1; m >>= 1) ss += __shfl_xor(ss, m, 32);
        if (lane == 0) vnormS[lrow] = ss;
        v4h h4;
        h4[0] = (_Float16)v4.x; h4[1] = (_Float16)v4.y;
        h4[2] = (_Float16)v4.z; h4[3] = (_Float16)v4.w;
        *(v4h*)(&aH[lrow][lane * 4]) = h4;
    }

    // ---- Async prefetch of codebook chunk 0 into buffer 0
    // chunk = 64 codes x 256B; thread covers 8 x 16B segments, seg = tid+128*i
    #pragma unroll
    for (int i = 0; i < SEGS_PER_THREAD; ++i) {
        const int      seg  = tid + 128 * i;
        const int      row  = seg >> 4;
        const int      colb = (seg & 15) * 16;
        const uint32_t lofs = bbase + (uint32_t)row * (AH_PITCH * 2) + colb;
        const uint64_t gadr = cb16b + (uint64_t)row * (K_DIM * 2) + colb;
        asm volatile("global_load_async_to_lds_b128 %0, %1, off"
                     :: "v"(lofs), "v"(gadr) : "memory");
    }

    __syncthreads();

    // ---- A fragments (ISA 16-bit A layout)
    v16h afrag[4];
    {
        const _Float16* ar = &aH[w * 16 + l16][0];
        #pragma unroll
        for (int kt = 0; kt < 4; ++kt) {
            union { v16h v; v8h p[2]; } u;
            u.p[0] = *(const v8h*)(ar + kt * 32 + hhalf * 8);
            u.p[1] = *(const v8h*)(ar + kt * 32 + hhalf * 8 + 16);
            afrag[kt] = u.v;
        }
    }

    float vn[8];
    #pragma unroll
    for (int j = 0; j < 8; ++j)
        vn[j] = vnormS[w * 16 + hhalf * 8 + j];

    float minv[8]; int mini[8];
    #pragma unroll
    for (int j = 0; j < 8; ++j) { minv[j] = 3.4e38f; mini[j] = 0; }

    // ---- Chunked N sweep: 16 chunks of 64 codes, double-buffered in LDS
    for (int ch = 0; ch < N_CHUNKS; ++ch) {
        // prefetch next chunk into the other buffer while this one computes
        if (ch + 1 < N_CHUNKS) {
            const uint32_t bufo = bbase + (uint32_t)((ch + 1) & 1) * bstride;
            const uint64_t gch  = cb16b
                + (uint64_t)(ch + 1) * (CHUNK_CODES * K_DIM * 2);
            #pragma unroll
            for (int i = 0; i < SEGS_PER_THREAD; ++i) {
                const int      seg  = tid + 128 * i;
                const int      row  = seg >> 4;
                const int      colb = (seg & 15) * 16;
                const uint32_t lofs = bufo + (uint32_t)row * (AH_PITCH * 2) + colb;
                const uint64_t gadr = gch + (uint64_t)row * (K_DIM * 2) + colb;
                asm volatile("global_load_async_to_lds_b128 %0, %1, off"
                             :: "v"(lofs), "v"(gadr) : "memory");
            }
            // wait for *this* chunk's 8 copies; next chunk's 8 stay in flight
            asm volatile("s_wait_asynccnt 0x8" ::: "memory");
        } else {
            asm volatile("s_wait_asynccnt 0x0" ::: "memory");
        }
        __syncthreads();   // all waves' copies for chunk `ch` visible

        const _Float16* bbuf = &bS[ch & 1][0][0];
        const int cb0 = ch * CHUNK_CODES;

        #pragma unroll
        for (int t = 0; t < 4; ++t) {           // 4 n-tiles of 16 codes
            const int nb = cb0 + t * 16;
            const _Float16* br = bbuf + (size_t)(t * 16 + l16) * AH_PITCH;
            v8f c = {0.f, 0.f, 0.f, 0.f, 0.f, 0.f, 0.f, 0.f};
            #pragma unroll
            for (int kt = 0; kt < 4; ++kt) {
                union { v16h v; v8h p[2]; } u;
                u.p[0] = *(const v8h*)(br + kt * 32 + hhalf * 8);
                u.p[1] = *(const v8h*)(br + kt * 32 + hhalf * 8 + 16);
                c = __builtin_amdgcn_wmma_f32_16x16x32_f16(
                        false, afrag[kt], false, u.v, (short)0, c, false, false);
            }
            const float cn   = cnorm[nb + l16];
            const int   nidx = nb + l16;
            #pragma unroll
            for (int j = 0; j < 8; ++j) {
                const float dist = vn[j] - 2.0f * c[j] + cn;
                if (dist < minv[j]) { minv[j] = dist; mini[j] = nidx; }
            }
        }
        __syncthreads();   // buffer ch&1 free for reuse (chunk ch+2)
    }

    // ---- argmin reduction across the 16 lanes holding different N
    #pragma unroll
    for (int j = 0; j < 8; ++j) {
        #pragma unroll
        for (int m = 1; m < 16; m <<= 1) {
            const float ov = __shfl_xor(minv[j], m, 16);
            const int   oi = __shfl_xor(mini[j], m, 16);
            if (ov < minv[j] || (ov == minv[j] && oi < mini[j])) {
                minv[j] = ov; mini[j] = oi;
            }
        }
    }
    if (l16 == 0) {
        #pragma unroll
        for (int j = 0; j < 8; ++j) {
            const int r = w * 16 + hhalf * 8 + j;
            zsh[r]   = mini[j];
            errsh[r] = fmaxf(minv[j], 0.0f);
        }
    }
    __syncthreads();

    // ---- write z / errs2; deterministic block partial of mask*errs2
    if (tid < ROWS_PER_BLOCK) {
        const long row = rowbase + tid;
        zout[row] = zsh[tid];
        eout[row] = errsh[tid];
        red[tid]  = errsh[tid] * mask[row];
    } else {
        red[tid] = 0.0f;
    }
    __syncthreads();
    for (int off = 64; off >= 1; off >>= 1) {
        if (tid < off) red[tid] += red[tid + off];
        __syncthreads();
    }
    if (tid == 0) partials[blockIdx.x] = red[0];

    // ---- vecs_hat = (c_z + v) - v, float4-vectorized
    {
        const int rg = tid >> 5;              // 0..3
        const int c4 = (tid & 31) * 4;        // column group
        for (int r = rg; r < ROWS_PER_BLOCK; r += 4) {
            const int  zr  = zsh[r];
            const long row = rowbase + r;
            const float4 cc = *(const float4*)(cb32 + (size_t)zr * K_DIM + c4);
            const float4 vv = *(const float4*)(vecs + row * K_DIM + c4);
            float4 o;
            o.x = (cc.x + vv.x) - vv.x;
            o.y = (cc.y + vv.y) - vv.y;
            o.z = (cc.z + vv.z) - vv.z;
            o.w = (cc.w + vv.w) - vv.w;
            *(float4*)(hat + row * K_DIM + c4) = o;
        }
    }
}

// ---------------------------------------------------------------------------
// Kernel 3: fixed-order reduction of per-block partials -> l_commit scalar
// ---------------------------------------------------------------------------
__global__ __launch_bounds__(256) void vq_commit_kernel(
    const float* __restrict__ partials, float* __restrict__ out)
{
    __shared__ float red[256];
    const int t = threadIdx.x;
    float s = 0.0f;
    for (int i = t; i < N_ROWS / ROWS_PER_BLOCK; i += 256) s += partials[i];
    red[t] = s;
    __syncthreads();
    for (int off = 128; off >= 1; off >>= 1) {
        if (t < off) red[t] += red[t + off];
        __syncthreads();
    }
    if (t == 0) out[0] = red[0] * (1.0f / (float)N_ROWS);
}

// ---------------------------------------------------------------------------
extern "C" void kernel_launch(void* const* d_in, const int* in_sizes, int n_in,
                              void* d_out, int out_size, void* d_ws, size_t ws_size,
                              hipStream_t stream)
{
    const float* vecs = (const float*)d_in[0];   // [8,1,8192,128] f32
    const float* mask = (const float*)d_in[1];   // [8,8192] f32

    float*     cb32     = (float*)d_ws;
    _Float16*  cb16     = (_Float16*)(cb32 + S_CODES * K_DIM);
    float*     cnorm    = (float*)(cb16 + S_CODES * K_DIM);
    float*     partials = cnorm + S_CODES;

    float* out      = (float*)d_out;
    float* hat      = out;
    int*   zout     = (int*)(out + (size_t)N_ROWS * K_DIM);
    float* eout     = out + (size_t)N_ROWS * K_DIM + N_ROWS;
    float* lcommit  = out + (size_t)N_ROWS * K_DIM + 2 * (size_t)N_ROWS;

    vq_codebook_kernel<<<S_CODES, 128, 0, stream>>>(cb32, cb16, cnorm);
    vq_main_kernel<<<N_ROWS / ROWS_PER_BLOCK, 128, 0, stream>>>(
        vecs, cb16, cb32, cnorm, mask, hat, zout, eout, partials);
    vq_commit_kernel<<<1, 256, 0, stream>>>(partials, lcommit);
}
